// FeatureEmbedding_66967130079884
// MI455X (gfx1250) — matrix operand order repaired and checked by reading, not compiled, not run on hardware
//
#include <hip/hip_runtime.h>

typedef __attribute__((ext_vector_type(16))) __bf16 v16bf;
typedef __attribute__((ext_vector_type(8)))  float  v8f;

__device__ __forceinline__ unsigned short f2bf(float f) {
  unsigned int u = __float_as_uint(f);
  unsigned int r = u + 0x7FFFu + ((u >> 16) & 1u);   // round-to-nearest-even
  return (unsigned short)(r >> 16);
}

// ---------------------------------------------------------------------------
// Repack input: x [N=384][Cin=5][64][64] f32 (NCHW)  ->  xpad NHWC bf16, C=32
// (channels 5..31 zero). One thread per pixel writes 32 bf16 = 64B.
// ---------------------------------------------------------------------------
__global__ void repack_input(const float* __restrict__ x,
                             unsigned short* __restrict__ xp, int total) {
  int gid = blockIdx.x * blockDim.x + threadIdx.x;
  if (gid >= total) return;
  int n = gid >> 12;          // /4096
  int p = gid & 4095;
  const float* src = x + (size_t)n * 5 * 4096 + p;
  unsigned short b[5];
#pragma unroll
  for (int c = 0; c < 5; ++c) b[c] = f2bf(src[(size_t)c * 4096]);
  unsigned int d0 = b[0] | ((unsigned int)b[1] << 16);
  unsigned int d1 = b[2] | ((unsigned int)b[3] << 16);
  unsigned int d2 = b[4];
  uint4* dst = (uint4*)(xp + (size_t)gid * 32);
  dst[0] = make_uint4(d0, d1, d2, 0u);
  dst[1] = make_uint4(0u, 0u, 0u, 0u);
  dst[2] = make_uint4(0u, 0u, 0u, 0u);
  dst[3] = make_uint4(0u, 0u, 0u, 0u);
}

// ---------------------------------------------------------------------------
// Repack weights [Cout][CinReal][3][3] f32 -> bf16 WMMA B-fragments.
// Fragment = 32x16 (KxN) bf16 in per-lane order: lane L holds column
// N = L&15, K half (L>=16 ? 16..31 : 0..15), K pairs packed in dwords.
// One thread per (fragment, lane) writes 32 contiguous bytes.
// frag index = (cinblk*9 + tap)*NT + tile.
// ---------------------------------------------------------------------------
__global__ void repack_weights(const float* __restrict__ w, uint4* __restrict__ out,
                               int CinReal, int NT, int CB) {
  int gid = blockIdx.x * blockDim.x + threadIdx.x;
  int nfrag = CB * 9 * NT;
  if (gid >= nfrag * 32) return;
  int frag = gid >> 5;
  int lane = gid & 31;
  int blk  = frag / (9 * NT);
  int rem  = frag % (9 * NT);
  int tap  = rem / NT;
  int tile = rem % NT;
  int ky = tap / 3, kx = tap % 3;
  int nCh   = tile * 16 + (lane & 15);
  int kbase = (lane >> 4) * 16;
  unsigned int dw[8];
#pragma unroll
  for (int d = 0; d < 8; ++d) {
    int cin0 = blk * 32 + kbase + 2 * d;
    unsigned short lo = 0, hi = 0;
    if (cin0 < CinReal)
      lo = f2bf(w[(((size_t)nCh * CinReal + cin0) * 3 + ky) * 3 + kx]);
    if (cin0 + 1 < CinReal)
      hi = f2bf(w[(((size_t)nCh * CinReal + cin0 + 1) * 3 + ky) * 3 + kx]);
    dw[d] = lo | ((unsigned int)hi << 16);
  }
  uint4* dst = out + (size_t)frag * 64 + lane * 2;
  dst[0] = make_uint4(dw[0], dw[1], dw[2], dw[3]);
  dst[1] = make_uint4(dw[4], dw[5], dw[6], dw[7]);
}

// ---------------------------------------------------------------------------
// Implicit-GEMM 3x3 conv, bf16 WMMA, f32 accumulate.
// Grid: (64 rows, 384 images). Block: 256 threads = 8 waves.
// Wave mapping: sp = wave&1 -> pixel strips {sp*32 .. sp*32+31} (2x16),
//               tg = wave>>1 -> cout tiles {tg*TPW2 .. tg*TPW2+TPW2-1},
// TPW2 = NT/4, so each wave holds 2*TPW2 accumulator tiles and reuses both
// A fragments (x TPW2) and B fragments (x 2 strips).
// Input NHWC bf16 (C = CB*32), staged to LDS with GLOBAL_LOAD_ASYNC_TO_LDS.
// Output: NHWC bf16, or (FINAL) NCHW f32 + positional encoding.
// ---------------------------------------------------------------------------
template <int CB, int NT, bool FINAL>
__global__ __launch_bounds__(256)
void conv3x3_wmma(const unsigned short* __restrict__ in,
                  const uint4* __restrict__ wfrag,
                  const float* __restrict__ bias,
                  void* __restrict__ outp) {
  constexpr int CIN  = CB * 32;
  constexpr int COUT = NT * 16;
  constexpr int TPW2 = NT / 4;      // cout tiles per wave
  constexpr int CSTR = 40;          // shorts per (row,col) cell: 32 data + 8 pad
                                    // 80B stride -> conflict-free ds_load_b128

  const int h    = blockIdx.x;
  const int n    = blockIdx.y;
  const int tid  = threadIdx.x;
  const int wave = tid >> 5;
  const int lane = tid & 31;
  const int sp   = wave & 1;        // strip pair: pixels sp*32 .. sp*32+31
  const int tg   = wave >> 1;       // cout tile group (0..3)
  const int lhalf = lane >> 4;      // 0|1
  const int l16   = lane & 15;

  __shared__ unsigned short s_in[3 * 66 * CSTR];   // halo tile, bf16
  __shared__ unsigned short s_out[8 * 16 * 64];    // per-wave transpose buffer

  v8f acc[2 * TPW2];
#pragma unroll
  for (int t = 0; t < 2 * TPW2; ++t)
#pragma unroll
    for (int r = 0; r < 8; ++r) acc[t][r] = 0.0f;

  const size_t img_base = (size_t)n * 4096;

  // ---- zero-fill halo cells once (async loads below only touch in-range) ---
  for (int idx = tid; idx < 3 * 66 * 4; idx += 256) {
    int row  = idx / (66 * 4);
    int rem  = idx - row * (66 * 4);
    int col  = rem >> 2;
    int part = rem & 3;
    int hh = h + row - 1;
    int ww = col - 1;
    if (!(hh >= 0 && hh < 64 && ww >= 0 && ww < 64))
      *(uint4*)(&s_in[(row * 66 + col) * CSTR + part * 8]) =
          make_uint4(0u, 0u, 0u, 0u);
  }

  for (int cb = 0; cb < CB; ++cb) {
    __syncthreads();
    // ---- async stage: rows h-1..h+1, cols -1..64, 32 channels -> LDS ------
    for (int idx = tid; idx < 3 * 66 * 4; idx += 256) {
      int row  = idx / (66 * 4);
      int rem  = idx - row * (66 * 4);
      int col  = rem >> 2;
      int part = rem & 3;                        // 16B chunk within 64B
      int hh = h + row - 1;
      int ww = col - 1;
      if (hh >= 0 && hh < 64 && ww >= 0 && ww < 64) {
        const unsigned short* gp =
            in + ((img_base + (size_t)hh * 64 + ww) * CIN + cb * 32 + part * 8);
        unsigned lds_off = (unsigned)(unsigned long long)(
            &s_in[(row * 66 + col) * CSTR + part * 8]);
        asm volatile("global_load_async_to_lds_b128 %0, %1, off"
                     :
                     : "v"(lds_off), "v"(gp)
                     : "memory");
      }
    }
    asm volatile("s_wait_asynccnt 0" ::: "memory");
    __syncthreads();

#pragma unroll
    for (int tap = 0; tap < 9; ++tap) {
      const int ty = tap / 3, tx = tap % 3;
      union Frag { uint4 q[2]; v16bf v; };
      Frag A[2], Bm[TPW2];
      // A fragments: 2 strips x (16 pixels x 32 cin), 2x ds_load_b128 each
#pragma unroll
      for (int si = 0; si < 2; ++si) {
        int col   = sp * 32 + si * 16 + l16 + tx;   // staged col (halo +1)
        int abase = (ty * 66 + col) * CSTR + (lhalf << 4);
        A[si].q[0] = *(const uint4*)(&s_in[abase]);
        A[si].q[1] = *(const uint4*)(&s_in[abase + 8]);
      }
      // B fragments: TPW2 tiles, 2x global_load_b128 each (L0/L2 resident)
#pragma unroll
      for (int ti = 0; ti < TPW2; ++ti) {
        const uint4* wsrc =
            wfrag + ((size_t)((cb * 9 + tap) * NT + tg * TPW2 + ti)) * 64 +
            lane * 2;
        Bm[ti].q[0] = wsrc[0];
        Bm[ti].q[1] = wsrc[1];
      }
#pragma unroll
      for (int si = 0; si < 2; ++si)
#pragma unroll
        for (int ti = 0; ti < TPW2; ++ti)
          acc[si * TPW2 + ti] = __builtin_amdgcn_wmma_f32_16x16x32_bf16(
              false, A[si].v, false, Bm[ti].v, (short)0, acc[si * TPW2 + ti],
              false, false);
    }
  }

  // ---------------------------- epilogue ----------------------------------
  if (FINAL) {
    float* out = (float*)outp;
    const int t_time = n % 12;
#pragma unroll
    for (int si = 0; si < 2; ++si) {
#pragma unroll
      for (int ti = 0; ti < TPW2; ++ti) {
        int cout = tg * TPW2 * 16 + ti * 16 + l16;
        float bv = bias[cout];
        int de = cout & ~1;
        float arg = (float)t_time * __expf(-((float)de * (1.0f / 128.0f)) *
                                           9.210340371976184f);  // ln(10000)
        float pe = (cout & 1) ? __cosf(arg) : __sinf(arg);
        float vals[8];
        const v8f a = acc[si * TPW2 + ti];
#pragma unroll
        for (int r = 0; r < 8; ++r) {
          float v = a[r] + bv;
          v = (v >= 0.0f) ? v : 0.01f * v;       // LeakyReLU
          vals[r] = v + pe;
        }
        size_t base = ((size_t)n * COUT + cout) * 4096 + (size_t)h * 64 +
                      sp * 32 + si * 16 + (lhalf << 3);
        *(float4*)(out + base)     = make_float4(vals[0], vals[1], vals[2], vals[3]);
        *(float4*)(out + base + 4) = make_float4(vals[4], vals[5], vals[6], vals[7]);
      }
    }
  } else {
    unsigned short* out = (unsigned short*)outp;
    // transpose through LDS: wave region = 32 pixels x TPW2*16 channels
#pragma unroll
    for (int si = 0; si < 2; ++si) {
#pragma unroll
      for (int ti = 0; ti < TPW2; ++ti) {
        int cout_l = ti * 16 + l16;
        float bv = bias[tg * TPW2 * 16 + cout_l];
        const v8f a = acc[si * TPW2 + ti];
#pragma unroll
        for (int r = 0; r < 8; ++r) {
          float v = a[r] + bv;
          v = (v >= 0.0f) ? v : 0.01f * v;
          int pixel_local = si * 16 + r + (lhalf << 3);
          s_out[wave * 1024 + pixel_local * (TPW2 * 16) + cout_l] = f2bf(v);
        }
      }
    }
    __syncthreads();
    // coalesced NHWC bf16 store: 16B chunks
#pragma unroll
    for (int j = 0; j < 2 * TPW2; ++j) {
      int c           = lane + 32 * j;
      int pixel_local = c / (2 * TPW2);
      int within      = c % (2 * TPW2);           // 8-short units
      uint4 v = *(const uint4*)(&s_out[wave * 1024 +
                                       pixel_local * (TPW2 * 16) + within * 8]);
      size_t base = (img_base + (size_t)h * 64 + sp * 32 + pixel_local) * COUT +
                    tg * TPW2 * 16 + within * 8;
      *(uint4*)(out + base) = v;
    }
  }
}

// ---------------------------------------------------------------------------
extern "C" void kernel_launch(void* const* d_in, const int* in_sizes, int n_in,
                              void* d_out, int out_size, void* d_ws, size_t ws_size,
                              hipStream_t stream) {
  const float* x  = (const float*)d_in[0];
  const float* w0 = (const float*)d_in[1];
  const float* b0 = (const float*)d_in[2];
  const float* w1 = (const float*)d_in[3];
  const float* b1 = (const float*)d_in[4];
  const float* w2 = (const float*)d_in[5];
  const float* b2 = (const float*)d_in[6];
  const float* w3 = (const float*)d_in[7];
  const float* b3 = (const float*)d_in[8];

  char* ws = (char*)d_ws;
  size_t off = 0;
  auto take = [&](size_t bytes) {
    size_t o = off;
    off = (off + bytes + 255) & ~(size_t)255;
    return o;
  };
  const size_t NP = 384ull * 4096ull;  // images * pixels
  unsigned short* xpad = (unsigned short*)(ws + take(NP * 32 * 2));
  unsigned short* buf0 = (unsigned short*)(ws + take(NP * 64 * 2));
  unsigned short* buf1 = (unsigned short*)(ws + take(NP * 128 * 2));
  unsigned short* buf2 = (unsigned short*)(ws + take(NP * 128 * 2));
  uint4* wf0 = (uint4*)(ws + take((size_t)1 * 9 * 4 * 1024));
  uint4* wf1 = (uint4*)(ws + take((size_t)2 * 9 * 8 * 1024));
  uint4* wf2 = (uint4*)(ws + take((size_t)4 * 9 * 8 * 1024));
  uint4* wf3 = (uint4*)(ws + take((size_t)4 * 9 * 8 * 1024));
  (void)ws_size; (void)in_sizes; (void)n_in; (void)out_size;

  repack_input<<<(int)((NP + 255) / 256), 256, 0, stream>>>(x, xpad, (int)NP);
  repack_weights<<<(1 * 9 * 4 * 32 + 255) / 256, 256, 0, stream>>>(w0, wf0, 5, 4, 1);
  repack_weights<<<(2 * 9 * 8 * 32 + 255) / 256, 256, 0, stream>>>(w1, wf1, 64, 8, 2);
  repack_weights<<<(4 * 9 * 8 * 32 + 255) / 256, 256, 0, stream>>>(w2, wf2, 128, 8, 4);
  repack_weights<<<(4 * 9 * 8 * 32 + 255) / 256, 256, 0, stream>>>(w3, wf3, 128, 8, 4);

  dim3 grid(64, 384);  // (row h, image n)
  conv3x3_wmma<1, 4, false><<<grid, 256, 0, stream>>>(xpad, wf0, b0, buf0);
  conv3x3_wmma<2, 8, false><<<grid, 256, 0, stream>>>(buf0, wf1, b1, buf1);
  conv3x3_wmma<4, 8, false><<<grid, 256, 0, stream>>>(buf1, wf2, b2, buf2);
  conv3x3_wmma<4, 8, true ><<<grid, 256, 0, stream>>>(buf2, wf3, b3, d_out);
}
